// PolymerGNN_IV_65609920413829
// MI455X (gfx1250) — matrix operand order repaired and checked by reading, not compiled, not run on hardware
//
#include <hip/hip_runtime.h>
#include <math.h>

// ---------------------------------------------------------------------------
// PolymerGNN_IV on MI455X (gfx1250): CSR-based graph aggregation (no feature
// atomics), f16 WMMA GEMMs (v_wmma_f32_16x16x32_f16) with f32 accumulate and
// fragment-swizzled operands (contiguous b128 LDS fragment loads), async
// global->LDS staging of weight tiles (GLOBAL_LOAD_ASYNC_TO_LDS_B128),
// radix-select top-K with stable tie-break, encoded-uint atomicMax pooling.
// ---------------------------------------------------------------------------

#define HH 256            // hidden channels (fixed by model)
#define NWAVE 8           // waves per 256-thread block

typedef _Float16 h16;
typedef __attribute__((ext_vector_type(16))) _Float16 v16h;
typedef __attribute__((ext_vector_type(8)))  float    v8f;

__device__ __forceinline__ float wave_sum(float v){
#pragma unroll
  for (int m = 16; m > 0; m >>= 1) v += __shfl_xor(v, m, 32);
  return v;
}
__device__ __forceinline__ float wave_max(float v){
#pragma unroll
  for (int m = 16; m > 0; m >>= 1) v = fmaxf(v, __shfl_xor(v, m, 32));
  return v;
}
// order-preserving float<->uint (for radix select + uint atomicMax)
__device__ __forceinline__ unsigned enc_f(float f){
  unsigned u = __float_as_uint(f);
  return (u >> 31) ? ~u : (u | 0x80000000u);
}
__device__ __forceinline__ float dec_f(unsigned u){
  return (u >> 31) ? __uint_as_float(u ^ 0x80000000u) : __uint_as_float(~u);
}
__device__ __forceinline__ float lrelu02(float e){ return e >= 0.f ? e : 0.2f * e; }

// ------------------------------- tiny utils --------------------------------
__global__ void pg_zero_u32(unsigned* p, int n){
  int i = blockIdx.x * blockDim.x + threadIdx.x;
  for (; i < n; i += gridDim.x * blockDim.x) p[i] = 0u;
}
__global__ void pg_copy_i32(const int* in, int* out, int n){
  int i = blockIdx.x * blockDim.x + threadIdx.x;
  for (; i < n; i += gridDim.x * blockDim.x) out[i] = in[i];
}

// Convert weight f32 [K,256] row-major -> f16 in WMMA *fragment order*:
// per 32-row K-chunk: [ntile(16)][lane(32)][t(16)] with
//   lane = (c&15) + ((r>>4)<<4), t = r&15   (r = k&31, ntile = c>>4)
// so each wave lane's B fragment is 16 contiguous halfs (32B).
__global__ void pg_wswz_f16(const float* __restrict__ in, h16* __restrict__ out, int n){
  int i = blockIdx.x * blockDim.x + threadIdx.x;
  for (; i < n; i += gridDim.x * blockDim.x){
    int k = i >> 8, c = i & 255;
    int chunk = k >> 5, r = k & 31;
    int ntile = c >> 4;
    int lane = (c & 15) + ((r >> 4) << 4);
    int t = r & 15;
    int dst = ((chunk * 16 + ntile) * 32 + lane) * 16 + t;
    out[dst] = (h16)in[i];
  }
}

// ------------------------------- CSR build ---------------------------------
__global__ void pg_count_deg(const int* __restrict__ dst, int E, int* __restrict__ deg){
  int i = blockIdx.x * blockDim.x + threadIdx.x;
  for (; i < E; i += gridDim.x * blockDim.x) atomicAdd(&deg[dst[i]], 1);
}
__global__ void pg_fill_csr(const int* __restrict__ src, const int* __restrict__ dst, int E,
                            int* __restrict__ pos, int* __restrict__ csr_src){
  int i = blockIdx.x * blockDim.x + threadIdx.x;
  for (; i < E; i += gridDim.x * blockDim.x){
    int slot = atomicAdd(&pos[dst[i]], 1);
    csr_src[slot] = src[i];
  }
}

// exclusive prefix scan: scan1 (per 256-block) -> scan2 (block sums, <=512)
// -> scan3 (add offsets)
__global__ void pg_scan1(const int* __restrict__ in, int n, int* __restrict__ out,
                         int* __restrict__ bsums){
  __shared__ int sh[256];
  int t = threadIdx.x, g = blockIdx.x * 256 + t;
  int v = (g < n) ? in[g] : 0;
  sh[t] = v; __syncthreads();
  for (int off = 1; off < 256; off <<= 1){
    int u = (t >= off) ? sh[t - off] : 0;
    __syncthreads();
    sh[t] += u;
    __syncthreads();
  }
  if (g < n) out[g] = sh[t] - v;             // exclusive within block
  if (t == 255) bsums[blockIdx.x] = sh[t];   // block total
}
__global__ void pg_scan2(int* bsums, int nb){ // single block of 512
  __shared__ int sh[512];
  int t = threadIdx.x;
  int v = (t < nb) ? bsums[t] : 0;
  sh[t] = v; __syncthreads();
  for (int off = 1; off < 512; off <<= 1){
    int u = (t >= off) ? sh[t - off] : 0;
    __syncthreads();
    sh[t] += u;
    __syncthreads();
  }
  if (t < nb) bsums[t] = sh[t] - v;          // exclusive block offsets
}
__global__ void pg_scan3(int* out, int n, const int* bsums){
  int g = blockIdx.x * 256 + threadIdx.x;
  if (g < n) out[g] += bsums[blockIdx.x];
}

// --------------------------- WMMA GEMM (N=256) -----------------------------
// C[M,256] = A1[M,K1]@B1 (+ A2[M,K2]@B2) (+ bias). A is f32, converted to
// fragment-swizzled f16 while staging to LDS. B is pre-swizzled f16 (above),
// so each 16KB K-chunk is staged verbatim with GLOBAL_LOAD_ASYNC_TO_LDS_B128.
// Block: 16 rows x 256 cols, 8 waves, each wave two 16x16 n-tiles.
// In-place safe when C aliases A1/A2 (row-wise dependence, block-local rows).
__global__ __launch_bounds__(256) void pg_gemm_wmma(
    const float* __restrict__ A1, const h16* __restrict__ B1,
    const float* __restrict__ A2, const h16* __restrict__ B2,
    const float* __restrict__ bias, float* __restrict__ C,
    int M, int K1, int K2)
{
  __shared__ h16 AsF[16 * 32];    // A chunk, fragment order [lane][t]
  __shared__ h16 BsF[16 * 32 * 16]; // B chunk, fragment order [ntile][lane][t]
  const int tid = threadIdx.x;
  const int wid = tid >> 5, lane = tid & 31;
  const int group = lane >> 4, l16 = lane & 15;
  const int m0 = blockIdx.x * 16;
  const unsigned ldsB0 = (unsigned)(size_t)(void*)&BsF[0]; // LDS byte offset

  v8f acc[2];
#pragma unroll
  for (int i = 0; i < 8; ++i){ acc[0][i] = 0.f; acc[1][i] = 0.f; }

  for (int pass = 0; pass < 2; ++pass){
    const float* Ap = pass ? A2 : A1;
    const h16*   Bp = pass ? B2 : B1;
    const int    K  = pass ? K2 : K1;
    if (Ap == nullptr) continue;
    for (int kc = 0; kc < K; kc += 32){
      __syncthreads();
      { // stage A: 16x32 f32 -> f16, scattered into fragment order.
        // element (row m, chunk-K k): lane = m + 16*(k>>4),
        //                             t = k&15
        int e = tid * 2, row = e >> 5, col = e & 31;
        int gr = m0 + row;
        float a0 = 0.f, a1 = 0.f;
        if (gr < M){
          const float* pA = Ap + (size_t)gr * K + kc + col;
          a0 = pA[0]; a1 = pA[1];
        }
        int l0 = row + ((col) >> 4) * 16;
        int t0 = col & 15;
        AsF[l0 * 16 + t0] = (h16)a0;
        int c1 = col + 1;
        int l1 = row + (c1 >> 4) * 16;
        int t1 = c1 & 15;
        AsF[l1 * 16 + t1] = (h16)a1;
      }
      { // stage B: 16KB contiguous chunk, async global->LDS b128 (ASYNCcnt)
        const h16* srcB = Bp + ((size_t)(kc >> 5)) * 8192;
#pragma unroll
        for (int it = 0; it < 4; ++it){
          int idx16 = tid + it * 256;                       // b128 unit index
          unsigned lo = ldsB0 + (unsigned)idx16 * 16u;      // LDS byte addr
          unsigned long long ga =
              (unsigned long long)(size_t)srcB + (unsigned long long)idx16 * 16ull;
          asm volatile("global_load_async_to_lds_b128 %0, %1, off"
                       :: "v"(lo), "v"(ga) : "memory");
        }
        asm volatile("s_wait_asynccnt 0" ::: "memory");
      }
      __syncthreads();
      // fragments: one contiguous 32B LDS read per operand per lane
      v16h af  = *(const v16h*)&AsF[lane * 16];
      const int nt0 = wid * 2;
      v16h bf0 = *(const v16h*)&BsF[(nt0 * 32 + lane) * 16];
      v16h bf1 = *(const v16h*)&BsF[((nt0 + 1) * 32 + lane) * 16];
      acc[0] = __builtin_amdgcn_wmma_f32_16x16x32_f16(false, af, false, bf0,
                                                      (short)0, acc[0], false, false);
      acc[1] = __builtin_amdgcn_wmma_f32_16x16x32_f16(false, af, false, bf1,
                                                      (short)0, acc[1], false, false);
    }
  }
  // epilogue: acc VGPR r -> row m0+r+group*8, col n0+l16
#pragma unroll
  for (int nt = 0; nt < 2; ++nt){
    int n0 = wid * 32 + nt * 16;
#pragma unroll
    for (int r = 0; r < 8; ++r){
      int row = m0 + r + group * 8;
      if (row < M){
        int col = n0 + l16;
        float v = acc[nt][r];
        if (bias) v += bias[col];
        C[(size_t)row * HH + col] = v;
      }
    }
  }
}

// ---------------------- per-node dual dot (wave/node) ----------------------
__global__ __launch_bounds__(256) void pg_dot2(const float* __restrict__ X,
    const float* __restrict__ w1, const float* __restrict__ w2,
    float* __restrict__ o1, float* __restrict__ o2, int n){
  int wid = threadIdx.x >> 5, lane = threadIdx.x & 31;
  int i = blockIdx.x * NWAVE + wid;
  if (i >= n) return;
  const float* x = X + (size_t)i * HH;
  float s1 = 0.f, s2 = 0.f;
#pragma unroll
  for (int k = 0; k < 8; ++k){
    int c = lane + k * 32; float v = x[c];
    s1 += v * w1[c]; s2 += v * w2[c];
  }
  s1 = wave_sum(s1); s2 = wave_sum(s2);
  if (lane == 0){ o1[i] = s1; o2[i] = s2; }
}

// --------------------------- GAT aggregation -------------------------------
// out[i] = max_{j in in(i) U {i}} h[src_j]*alpha_j + bias (softmax over e)
__global__ __launch_bounds__(256) void pg_gat_aggregate(
    const float* __restrict__ h, const float* __restrict__ es,
    const float* __restrict__ ed, const int* __restrict__ offs,
    const int* __restrict__ deg, const int* __restrict__ csr_src,
    const float* __restrict__ bias, float* __restrict__ out, int n)
{
  int wid = threadIdx.x >> 5, lane = threadIdx.x & 31;
  int i = blockIdx.x * NWAVE + wid;
  if (i >= n) return;
  int base = offs[i], d = deg[i];
  float edi = ed[i];
  float eself = lrelu02(es[i] + edi);
  // pass 1: max logit
  float m = -INFINITY;
  for (int j = lane; j < d; j += 32){
    int s = csr_src[base + j];
    m = fmaxf(m, lrelu02(es[s] + edi));
  }
  m = fmaxf(wave_max(m), eself);
  // pass 2: sum exp
  float sm = 0.f;
  for (int j = lane; j < d; j += 32){
    int s = csr_src[base + j];
    sm += expf(lrelu02(es[s] + edi) - m);
  }
  sm = wave_sum(sm) + expf(eself - m);
  float inv = 1.0f / (sm + 1e-16f);
  // pass 3: feature max (8 channels per lane)
  float acc[8];
  float aself = expf(eself - m) * inv;
  const float* hi = h + (size_t)i * HH;
#pragma unroll
  for (int k = 0; k < 8; ++k) acc[k] = hi[lane + k * 32] * aself;
  for (int j = 0; j < d; ++j){
    int s = csr_src[base + j];
    float al = expf(lrelu02(es[s] + edi) - m) * inv;
    const float* hs = h + (size_t)s * HH;
#pragma unroll
    for (int k = 0; k < 8; ++k) acc[k] = fmaxf(acc[k], hs[lane + k * 32] * al);
  }
  float* oi = out + (size_t)i * HH;
#pragma unroll
  for (int k = 0; k < 8; ++k) oi[lane + k * 32] = acc[k] + bias[lane + k * 32];
}

// --------------------------- SAGE max-aggregate ----------------------------
__global__ __launch_bounds__(256) void pg_sage_agg(
    const float* __restrict__ X, const int* __restrict__ offs,
    const int* __restrict__ deg, const int* __restrict__ csr_src,
    float* __restrict__ out, int n)
{
  int wid = threadIdx.x >> 5, lane = threadIdx.x & 31;
  int i = blockIdx.x * NWAVE + wid;
  if (i >= n) return;
  int base = offs[i], d = deg[i];
  float acc[8];
#pragma unroll
  for (int k = 0; k < 8; ++k) acc[k] = -INFINITY;
  for (int j = 0; j < d; ++j){
    int s = csr_src[base + j];
    const float* xs = X + (size_t)s * HH;
#pragma unroll
    for (int k = 0; k < 8; ++k) acc[k] = fmaxf(acc[k], xs[lane + k * 32]);
  }
  float* oi = out + (size_t)i * HH;
#pragma unroll
  for (int k = 0; k < 8; ++k) oi[lane + k * 32] = (d > 0) ? acc[k] : 0.f;
}

// ------------------------------- BatchNorm ---------------------------------
__global__ __launch_bounds__(256) void pg_bn_stats(const float* __restrict__ X, int n,
                                                   float* __restrict__ gsum,
                                                   float* __restrict__ gsq){
  int t = threadIdx.x;
  float s = 0.f, q = 0.f;
  for (int r = blockIdx.x; r < n; r += gridDim.x){
    float v = X[(size_t)r * HH + t];
    s += v; q += v * v;
  }
  atomicAdd(&gsum[t], s);
  atomicAdd(&gsq[t], q);
}
__global__ void pg_bn_finalize(const float* gsum, const float* gsq, int n,
                               const float* g, const float* b,
                               float* scale, float* shift){
  int t = threadIdx.x;
  float mean = gsum[t] / (float)n;
  float var  = gsq[t] / (float)n - mean * mean;
  float sc = g[t] * rsqrtf(var + 1e-5f);
  scale[t] = sc;
  shift[t] = b[t] - mean * sc;
}
__global__ void pg_bn_apply_prelu(float* __restrict__ X, long n4,
                                  const float* __restrict__ scale,
                                  const float* __restrict__ shift,
                                  const float* __restrict__ aP){
  float a = aP[0];
  long i = (long)blockIdx.x * blockDim.x + threadIdx.x;
  for (; i < n4; i += (long)gridDim.x * blockDim.x){
    float4 v = ((float4*)X)[i];
    int c = (int)((i * 4) & (HH - 1));
    v.x = v.x * scale[c]     + shift[c];
    v.y = v.y * scale[c + 1] + shift[c + 1];
    v.z = v.z * scale[c + 2] + shift[c + 2];
    v.w = v.w * scale[c + 3] + shift[c + 3];
    v.x = v.x >= 0.f ? v.x : a * v.x;
    v.y = v.y >= 0.f ? v.y : a * v.y;
    v.z = v.z >= 0.f ? v.z : a * v.z;
    v.w = v.w >= 0.f ? v.w : a * v.w;
    ((float4*)X)[i] = v;
  }
}

// ------------------------- SAGPool score + top-K ---------------------------
__global__ void pg_score(const int* __restrict__ offs, const int* __restrict__ deg,
                         const int* __restrict__ csr_src, const float* __restrict__ rb,
                         const float* __restrict__ tb, const float* __restrict__ brel,
                         float* __restrict__ score, unsigned* __restrict__ ukey, int n){
  int i = blockIdx.x * blockDim.x + threadIdx.x;
  if (i >= n) return;
  int base = offs[i], d = deg[i];
  float s = 0.f;
  for (int j = 0; j < d; ++j) s += rb[csr_src[base + j]];
  float sc = s + brel[0] + tb[i];
  score[i] = sc;
  ukey[i] = enc_f(sc);
}
__global__ void pg_init_state(unsigned* state, unsigned Ksel){
  state[0] = 0u;     // prefix
  state[1] = Ksel;   // remaining
}
__global__ void pg_hist(const unsigned* __restrict__ ukey, int n,
                        const unsigned* __restrict__ state, int shift,
                        unsigned* __restrict__ hist){
  unsigned prefix = state[0];
  unsigned mask = (shift >= 24) ? 0u : (0xFFFFFFFFu << (shift + 8));
  int i = blockIdx.x * blockDim.x + threadIdx.x;
  for (; i < n; i += gridDim.x * blockDim.x){
    unsigned u = ukey[i];
    if ((u & mask) == prefix) atomicAdd(&hist[(u >> shift) & 0xFFu], 1u);
  }
}
__global__ void pg_pick(unsigned* state, const unsigned* hist, int shift){
  unsigned kRem = state[1];
  unsigned cum = 0; int bin = 0;
  for (int b = 255; b >= 0; --b){
    if (cum + hist[b] >= kRem){ bin = b; break; }
    cum += hist[b];
  }
  state[0] |= ((unsigned)bin) << shift;
  state[1] = kRem - cum;  // still needed within chosen bin
}
__global__ void pg_eqflag(const unsigned* ukey, const unsigned* state, int* flag, int n){
  int i = blockIdx.x * blockDim.x + threadIdx.x;
  if (i < n) flag[i] = (ukey[i] == state[0]) ? 1 : 0;
}
__global__ void pg_elig(const unsigned* __restrict__ ukey, const unsigned* __restrict__ state,
                        const int* __restrict__ eqrank, const float* __restrict__ score,
                        int* __restrict__ elig, float* __restrict__ wgt, int n){
  int i = blockIdx.x * blockDim.x + threadIdx.x;
  if (i >= n) return;
  unsigned T = state[0], kEq = state[1];
  unsigned u = ukey[i];
  elig[i] = (u > T) || (u == T && (unsigned)eqrank[i] < kEq);
  wgt[i] = tanhf(score[i]);
}
__global__ void pg_init_umax(unsigned* umax){
  umax[threadIdx.x] = enc_f(-INFINITY);
}
__global__ __launch_bounds__(256) void pg_pool_max(const float* __restrict__ X,
    const int* __restrict__ elig, const float* __restrict__ wgt,
    unsigned* __restrict__ umax, int n){
  int t = threadIdx.x;
  float local = -INFINITY;
  for (int r = blockIdx.x; r < n; r += gridDim.x){
    if (elig[r]) local = fmaxf(local, X[(size_t)r * HH + t] * wgt[r]);
  }
  atomicMax(&umax[t], enc_f(local));
}
__global__ void pg_decode(const unsigned* umax, float* out){
  out[threadIdx.x] = dec_f(umax[threadIdx.x]);
}

// --------------------------------- FC head ---------------------------------
__global__ __launch_bounds__(256) void pg_head(const float* Ae, const float* Ge,
    const float* addf, int na, const float* W1, const float* b1, const float* aP,
    const float* W2, const float* b2, float* out){
  __shared__ float z[2 * HH + 32];
  __shared__ float red[HH];
  int t = threadIdx.x;
  z[t] = Ae[t]; z[HH + t] = Ge[t];
  if (t < na) z[2 * HH + t] = addf[t];
  __syncthreads();
  int ktot = 2 * HH + na;
  float h = 0.f;
  for (int j = 0; j < ktot; ++j) h += z[j] * W1[(size_t)j * HH + t];
  h += b1[t];
  float a = aP[0];
  h = h >= 0.f ? h : a * h;
  red[t] = h * W2[t];
  __syncthreads();
  for (int off = HH / 2; off > 0; off >>= 1){
    if (t < off) red[t] += red[t + off];
    __syncthreads();
  }
  if (t == 0) out[0] = expf(red[0] + b2[0]);
}

// ================================ launcher =================================
extern "C" void kernel_launch(void* const* d_in, const int* in_sizes, int n_in,
                              void* d_out, int out_size, void* d_ws, size_t ws_size,
                              hipStream_t stream){
  const int N  = in_sizes[0] / 64;       // 100000
  const int F  = in_sizes[5] / HH;       // 64
  const int E  = in_sizes[3] / 2;        // 400000
  const int NA = in_sizes[2];            // 8
  const unsigned Ksel = (unsigned)((N + 1) / 2);

  const float* A_x  = (const float*)d_in[0];
  const float* G_x  = (const float*)d_in[1];
  const float* addf = (const float*)d_in[2];
  const int*   A_ei = (const int*)d_in[3];
  const int*   G_ei = (const int*)d_in[4];
  const float* PA[16]; const float* PG[16];
  for (int i = 0; i < 16; ++i){ PA[i] = (const float*)d_in[5 + i]; PG[i] = (const float*)d_in[21 + i]; }
  const float* fc1W = (const float*)d_in[37];
  const float* fc1b = (const float*)d_in[38];
  const float* preluh = (const float*)d_in[39];
  const float* fc2W = (const float*)d_in[40];
  const float* fc2b = (const float*)d_in[41];

  // workspace bump allocator (256B aligned)
  char* p = (char*)d_ws;
  auto alloc = [&](size_t bytes) -> void* {
    void* r = (void*)p;
    p += (bytes + 255) & ~(size_t)255;
    return r;
  };
  float*    H1      = (float*)alloc((size_t)N * HH * 4);
  float*    H2      = (float*)alloc((size_t)N * HH * 4);
  h16*      Wf16a   = (h16*)alloc((size_t)HH * HH * 2);
  h16*      Wf16b   = (h16*)alloc((size_t)HH * HH * 2);
  float*    es_     = (float*)alloc((size_t)N * 4);
  float*    ed_     = (float*)alloc((size_t)N * 4);
  int*      deg     = (int*)alloc((size_t)N * 4);
  int*      offs    = (int*)alloc((size_t)N * 4);
  int*      pos     = (int*)alloc((size_t)N * 4);
  int*      csr_src = (int*)alloc((size_t)E * 4);
  int*      sbuf    = (int*)alloc(1024 * 4);
  float*    gstat   = (float*)alloc(512 * 4);        // gsum[256], gsq[256]
  float*    bnsc    = (float*)alloc(HH * 4);
  float*    bnsh    = (float*)alloc(HH * 4);
  float*    rbuf    = (float*)alloc((size_t)N * 4);
  float*    tbuf    = (float*)alloc((size_t)N * 4);
  float*    score   = (float*)alloc((size_t)N * 4);
  float*    wgt     = (float*)alloc((size_t)N * 4);
  unsigned* ukey    = (unsigned*)alloc((size_t)N * 4);
  int*      eqflag  = (int*)alloc((size_t)N * 4);
  int*      eqrank  = (int*)alloc((size_t)N * 4);
  int*      elig    = (int*)alloc((size_t)N * 4);
  unsigned* hist    = (unsigned*)alloc(256 * 4);
  unsigned* state   = (unsigned*)alloc(64);
  unsigned* umax    = (unsigned*)alloc(HH * 4);
  float*    Ae      = (float*)alloc(HH * 4);
  float*    Ge      = (float*)alloc(HH * 4);
  (void)ws_size; (void)n_in; (void)out_size;

  const int nbN = (N + 255) / 256;   // 1D N kernels
  const int nbE = (E + 255) / 256;
  const int nbW = (N + NWAVE - 1) / NWAVE;  // wave-per-node
  const int nbG = (N + 15) / 16;            // GEMM m-tiles
  const int nbScan = (N + 255) / 256;
  const long n4 = (long)N * HH / 4;

  auto scan_exclusive = [&](const int* in, int* out, int n){
    int nb = (n + 255) / 256;
    pg_scan1<<<nb, 256, 0, stream>>>(in, n, out, sbuf);
    pg_scan2<<<1, 512, 0, stream>>>(sbuf, nb);
    pg_scan3<<<nb, 256, 0, stream>>>(out, n, sbuf);
  };

  auto run_branch = [&](const float* x, const int* ei, const float* const* P, float* outvec){
    const int* src = ei;
    const int* dst = ei + E;
    // --- GAT: h = x @ W (WMMA, fragment-swizzled weights) ---
    pg_wswz_f16<<<(F * HH + 255) / 256, 256, 0, stream>>>(P[0], Wf16a, F * HH);
    pg_gemm_wmma<<<nbG, 256, 0, stream>>>(x, Wf16a, nullptr, nullptr, nullptr, H1, N, F, 0);
    // es = h@a_src, ed = h@a_dst
    pg_dot2<<<nbW, 256, 0, stream>>>(H1, P[1], P[2], es_, ed_, N);
    // --- CSR by dst ---
    pg_zero_u32<<<nbN, 256, 0, stream>>>((unsigned*)deg, N);
    pg_count_deg<<<nbE, 256, 0, stream>>>(dst, E, deg);
    scan_exclusive(deg, offs, N);
    pg_copy_i32<<<nbScan, 256, 0, stream>>>(offs, pos, N);
    pg_fill_csr<<<nbE, 256, 0, stream>>>(src, dst, E, pos, csr_src);
    // --- GAT softmax + max aggregation (+bias) -> H2 ---
    pg_gat_aggregate<<<nbW, 256, 0, stream>>>(H1, es_, ed_, offs, deg, csr_src, P[3], H2, N);
    // --- BN1 + PReLU (in place on H2) ---
    pg_zero_u32<<<2, 256, 0, stream>>>((unsigned*)gstat, 512);
    pg_bn_stats<<<512, 256, 0, stream>>>(H2, N, gstat, gstat + HH);
    pg_bn_finalize<<<1, 256, 0, stream>>>(gstat, gstat + HH, N, P[4], P[5], bnsc, bnsh);
    pg_bn_apply_prelu<<<2048, 256, 0, stream>>>(H2, n4, bnsc, bnsh, P[6]);
    // --- SAGE: agg = max_in(H2) -> H1 ; H2 = H1@Wl + H2@Wr + bl (in place) ---
    pg_sage_agg<<<nbW, 256, 0, stream>>>(H2, offs, deg, csr_src, H1, N);
    pg_wswz_f16<<<(HH * HH + 255) / 256, 256, 0, stream>>>(P[7], Wf16a, HH * HH);
    pg_wswz_f16<<<(HH * HH + 255) / 256, 256, 0, stream>>>(P[9], Wf16b, HH * HH);
    pg_gemm_wmma<<<nbG, 256, 0, stream>>>(H1, Wf16a, H2, Wf16b, P[8], H2, N, HH, HH);
    // --- BN2 + PReLU (in place on H2) ---
    pg_zero_u32<<<2, 256, 0, stream>>>((unsigned*)gstat, 512);
    pg_bn_stats<<<512, 256, 0, stream>>>(H2, N, gstat, gstat + HH);
    pg_bn_finalize<<<1, 256, 0, stream>>>(gstat, gstat + HH, N, P[10], P[11], bnsc, bnsh);
    pg_bn_apply_prelu<<<2048, 256, 0, stream>>>(H2, n4, bnsc, bnsh, P[12]);
    // --- SAGPool score: score[i] = sum_in r[src] + brel + t[i] ---
    pg_dot2<<<nbW, 256, 0, stream>>>(H2, P[13], P[15], rbuf, tbuf, N);
    pg_score<<<nbN, 256, 0, stream>>>(offs, deg, csr_src, rbuf, tbuf, P[14], score, ukey, N);
    // --- radix select K-th largest over encoded keys ---
    pg_init_state<<<1, 1, 0, stream>>>(state, Ksel);
    for (int shift = 24; shift >= 0; shift -= 8){
      pg_zero_u32<<<1, 256, 0, stream>>>(hist, 256);
      pg_hist<<<nbN, 256, 0, stream>>>(ukey, N, state, shift, hist);
      pg_pick<<<1, 1, 0, stream>>>(state, hist, shift);
    }
    // stable tie-break: rank among keys == threshold (lowest index wins)
    pg_eqflag<<<nbN, 256, 0, stream>>>(ukey, state, eqflag, N);
    scan_exclusive(eqflag, eqrank, N);
    pg_elig<<<nbN, 256, 0, stream>>>(ukey, state, eqrank, score, elig, wgt, N);
    // --- channelwise max over selected rows * tanh(score) ---
    pg_init_umax<<<1, 256, 0, stream>>>(umax);
    pg_pool_max<<<512, 256, 0, stream>>>(H2, elig, wgt, umax, N);
    pg_decode<<<1, 256, 0, stream>>>(umax, outvec);
  };

  run_branch(A_x, A_ei, PA, Ae);
  run_branch(G_x, G_ei, PG, Ge);

  pg_head<<<1, 256, 0, stream>>>(Ae, Ge, addf, NA, fc1W, fc1b, preluh, fc2W, fc2b,
                                 (float*)d_out);
}
